// INNLightGCNLinkPredictor_88768384074361
// MI455X (gfx1250) — compile-verified, matrix operand order checked by reading, not run on hardware
//
#include <hip/hip_runtime.h>

typedef __attribute__((ext_vector_type(16))) _Float16 v16h;
typedef __attribute__((ext_vector_type(8)))  float    v8f;

#define DIM  128
#define NEGS 128

__device__ __forceinline__ float softplus_f(float x) {
    // numerically stable softplus
    return fmaxf(x, 0.0f) + log1pf(expf(-fabsf(x)));
}

// One wave per row: Srho[row] = sum_k softplus(rho[row,k]), k=0..127.
// Lane loads a float4 (global_load_b128), reduces 4 in VALU, then 5-step
// xor-shuffle wave reduction (wave32).
__global__ void rho_rowsum_kernel(const float* __restrict__ rho,
                                  float* __restrict__ out, int nrows) {
    const int lane = threadIdx.x & 31;
    const int wave = threadIdx.x >> 5;
    const int row  = blockIdx.x * (blockDim.x >> 5) + wave;
    if (row >= nrows) return;                       // whole wave exits together
    const float4 v = *reinterpret_cast<const float4*>(rho + (size_t)row * DIM + lane * 4);
    float s = softplus_f(v.x) + softplus_f(v.y) + softplus_f(v.z) + softplus_f(v.w);
#pragma unroll
    for (int off = 16; off >= 1; off >>= 1) s += __shfl_xor(s, off, 32);
    if (lane == 0) out[row] = s;
}

// ---- shared epilogue: 16 rows/wave, per-lane f32 partial -> one WMMA ----
// A (16x32 f16): lane m slot0 = -partial, lane m slot1 = rad (low half only),
// lane m+16 slot0 = -partial. B = ones => D[m,*] = rad[m] - dist[m] = score.
__device__ __forceinline__ void wmma_reduce_store(float neg_partial, float rad,
                                                  int lane, float* __restrict__ o16) {
    v16h a = {};
    a[0] = (_Float16)neg_partial;
    a[1] = (lane >> 4) ? (_Float16)0.0f : (_Float16)rad;
    v16h bones;
#pragma unroll
    for (int i = 0; i < 16; ++i) bones[i] = (_Float16)1.0f;
    v8f c = {};
    v8f d = __builtin_amdgcn_wmma_f32_16x16x32_f16(false, a, false, bones,
                                                   (short)0, c, false, false);
    // lane 0: rows 0..7 ; lane 16: rows 8..15 (all N columns identical)
    if (lane == 0 || lane == 16) {
        float* o = o16 + ((lane == 16) ? 8 : 0);
        float4 lo = {d[0], d[1], d[2], d[3]};
        float4 hi = {d[4], d[5], d[6], d[7]};
        *reinterpret_cast<float4*>(o)     = lo;
        *reinterpret_cast<float4*>(o + 4) = hi;
    }
}

// One block (256 thr = 8 waves) per positive triplet b; wave w handles
// negatives [w*16, w*16+16). Lane L and L+16 split row L%16's 128 dims.
__global__ void neg_score_kernel(const int* __restrict__ pos_trip,
                                 const int* __restrict__ neg_trip,
                                 const float* __restrict__ ecen,
                                 const float* __restrict__ rcen,
                                 const float* __restrict__ srho,
                                 const float* __restrict__ srel,
                                 float* __restrict__ out) {
    __shared__ __align__(16) float rc_s[DIM];
    const int b    = blockIdx.x;
    const int tid  = threadIdx.x;
    const int lane = tid & 31;
    const int wave = tid >> 5;
    const int rel  = pos_trip[b * 3 + 1];

    if (tid < 32) {  // stage relation center row in LDS (512 B)
        const float4 r4 = *reinterpret_cast<const float4*>(rcen + (size_t)rel * DIM + tid * 4);
        *reinterpret_cast<float4*>(&rc_s[tid * 4]) = r4;
    }
    __syncthreads();

    const int m    = lane & 15;
    const int half = lane >> 4;
    const int n    = wave * 16 + m;
    const long tb  = ((long)b * NEGS + n) * 3;
    const int nh   = neg_trip[tb + 0];
    const int nt   = neg_trip[tb + 2];
    const float* __restrict__ hrow = ecen + (size_t)nh * DIM;
    const float* __restrict__ trow = ecen + (size_t)nt * DIM;

    float acc = 0.0f;
    const int kbase = half * 64;
#pragma unroll
    for (int kk = 0; kk < 64; kk += 4) {
        const int k = kbase + kk;
        const float4 h4 = *reinterpret_cast<const float4*>(hrow + k);
        const float4 t4 = *reinterpret_cast<const float4*>(trow + k);
        const float4 r4 = *reinterpret_cast<const float4*>(&rc_s[k]);
        acc += fabsf(h4.x + r4.x - t4.x);
        acc += fabsf(h4.y + r4.y - t4.y);
        acc += fabsf(h4.z + r4.z - t4.z);
        acc += fabsf(h4.w + r4.w - t4.w);
    }
    const float rad = srho[nh] + srho[nt] + srel[rel];
    wmma_reduce_store(-acc, rad, lane, out + (size_t)b * NEGS + wave * 16);
}

// 256 thr = 8 waves = 128 positive rows per block; grid = B/128.
__global__ void pos_score_kernel(const int* __restrict__ pos_trip,
                                 const float* __restrict__ ecen,
                                 const float* __restrict__ rcen,
                                 const float* __restrict__ srho,
                                 const float* __restrict__ srel,
                                 float* __restrict__ out) {
    const int tid  = threadIdx.x;
    const int lane = tid & 31;
    const int wave = tid >> 5;
    const int m    = lane & 15;
    const int half = lane >> 4;
    const int b    = blockIdx.x * 128 + wave * 16 + m;

    const int h = pos_trip[b * 3 + 0];
    const int r = pos_trip[b * 3 + 1];
    const int t = pos_trip[b * 3 + 2];
    const float* __restrict__ hrow = ecen + (size_t)h * DIM;
    const float* __restrict__ rrow = rcen + (size_t)r * DIM;
    const float* __restrict__ trow = ecen + (size_t)t * DIM;

    float acc = 0.0f;
    const int kbase = half * 64;
#pragma unroll
    for (int kk = 0; kk < 64; kk += 4) {
        const int k = kbase + kk;
        const float4 h4 = *reinterpret_cast<const float4*>(hrow + k);
        const float4 r4 = *reinterpret_cast<const float4*>(rrow + k);
        const float4 t4 = *reinterpret_cast<const float4*>(trow + k);
        acc += fabsf(h4.x + r4.x - t4.x);
        acc += fabsf(h4.y + r4.y - t4.y);
        acc += fabsf(h4.z + r4.z - t4.z);
        acc += fabsf(h4.w + r4.w - t4.w);
    }
    const float rad = srho[h] + srel[r] + srho[t];
    wmma_reduce_store(-acc, rad, lane, out + (size_t)blockIdx.x * 128 + wave * 16);
}

extern "C" void kernel_launch(void* const* d_in, const int* in_sizes, int n_in,
                              void* d_out, int out_size, void* d_ws, size_t ws_size,
                              hipStream_t stream) {
    const int*   pos  = (const int*)d_in[0];
    const int*   neg  = (const int*)d_in[1];
    const float* ecen = (const float*)d_in[2];
    const float* erho = (const float*)d_in[3];
    const float* rcen = (const float*)d_in[4];
    const float* rrho = (const float*)d_in[5];
    float* out = (float*)d_out;

    const int nent = in_sizes[2] / DIM;   // 100000
    const int nrel = in_sizes[4] / DIM;   // 500
    const int B    = in_sizes[0] / 3;     // 2048

    float* srho = (float*)d_ws;           // nent floats
    float* srel = srho + nent;            // nrel floats  (total ~402 KB << ws)

    // Pass 1: per-row softplus sums (exact radius-term collapse).
    rho_rowsum_kernel<<<(nent + 7) / 8, 256, 0, stream>>>(erho, srho, nent);
    rho_rowsum_kernel<<<(nrel + 7) / 8, 256, 0, stream>>>(rrho, srel, nrel);

    // Pass 2: scores. Same-stream launches give the needed ordering.
    pos_score_kernel<<<B / 128, 256, 0, stream>>>(pos, ecen, rcen, srho, srel, out);
    neg_score_kernel<<<B, 256, 0, stream>>>(pos, neg, ecen, rcen, srho, srel, out + B);
}
